// GnnModel_197568496161
// MI455X (gfx1250) — matrix operand order repaired and checked by reading, not compiled, not run on hardware
//
#include <hip/hip_runtime.h>
#include <hip/hip_bf16.h>

// CDNA5 WMMA vector types
typedef __attribute__((ext_vector_type(16))) __bf16 v16bf;
typedef __attribute__((ext_vector_type(8)))  float  v8f;

#define FDIM 128

// ---------- helpers ----------
__device__ __forceinline__ unsigned short f2bf(float f) {
    unsigned b = __float_as_uint(f);
    unsigned r = b + 0x7FFFu + ((b >> 16) & 1u);   // round-to-nearest-even
    return (unsigned short)(r >> 16);
}
// order-preserving float<->uint transform for atomicMax on signed floats
__device__ __forceinline__ unsigned fflip(float f) {
    unsigned b = __float_as_uint(f);
    return (b & 0x80000000u) ? ~b : (b | 0x80000000u);
}
__device__ __forceinline__ float funflip(unsigned u) {
    unsigned b = (u & 0x80000000u) ? (u & 0x7FFFFFFFu) : ~u;
    return __uint_as_float(b);
}

// ---------- init: zero agg/denom, emax floor ----------
__global__ void init_kernel(float* agg, float* denom, unsigned* emax, int n, int nf) {
    int i = blockIdx.x * blockDim.x + threadIdx.x;
    if (i < nf) agg[i] = 0.0f;
    if (i < n) { denom[i] = 0.0f; emax[i] = 0u; }
}

// ---------- f32 -> bf16 bits ----------
__global__ void cvt_kernel(const float* __restrict__ in, unsigned short* __restrict__ out, int n) {
    int i = blockIdx.x * blockDim.x + threadIdx.x;
    if (i < n) out[i] = f2bf(in[i]);
}

// ---------- generic wave-per-row dot: out[r] = A[r,:]·v + bias ----------
__global__ void rowdot_kernel(const float* __restrict__ A, const float* __restrict__ v,
                              const float* __restrict__ bias, float* __restrict__ out,
                              int nrows, int K) {
    int row  = blockIdx.x * (blockDim.x >> 5) + (threadIdx.x >> 5);
    if (row >= nrows) return;
    int lane = threadIdx.x & 31;
    const float* ar = A + (size_t)row * K;
    float sum = 0.0f;
    for (int k = lane; k < K; k += 32) sum += ar[k] * v[k];
    #pragma unroll
    for (int off = 16; off > 0; off >>= 1) sum += __shfl_xor(sum, off, 32);
    if (lane == 0) out[row] = sum + (bias ? bias[0] : 0.0f);
}

// ---------- bf16 WMMA GEMM: out = act(A[nrows,128] @ B[128,NCOLS] + bias + Cin) ----------
// Persistent blocks; B staged once per block into LDS; fragments software-pipelined.
template <int NCOLS>
__global__ void gemm_bf16_wmma_kernel(const unsigned short* __restrict__ A,
                                      const unsigned short* __restrict__ B,
                                      const float* __restrict__ bias,
                                      const float* __restrict__ Cin,
                                      float* __restrict__ outF,
                                      unsigned short* __restrict__ outBf,
                                      int nrows, int relu) {
    constexpr int NT = NCOLS / 16;
    // B staged in LDS, row-major, 16B-aligned (32KB for NCOLS=128, 16KB for 64)
    __shared__ uint4 Bs4[FDIM * NCOLS / 8];
    {
        const uint4* gb = (const uint4*)B;
        const int total = FDIM * NCOLS / 8;
        for (int i = threadIdx.x; i < total; i += blockDim.x) Bs4[i] = gb[i];
    }
    __syncthreads();
    const unsigned short* Bs = (const unsigned short*)Bs4;

    int lane    = threadIdx.x & 31;
    int wid     = threadIdx.x >> 5;
    int wpb     = blockDim.x >> 5;
    int halfsel = lane >> 4;                  // 0: lanes 0-15, 1: lanes 16-31
    int lr      = lane & 15;
    int ntiles  = (nrows + 15) >> 4;

    // persistent grid-stride loop over 16-row tiles (wave-uniform -> EXEC all-ones)
    for (int tile = blockIdx.x * wpb + wid; tile < ntiles; tile += gridDim.x * wpb) {
        int mbase = tile * 16;

        // seed accumulators: bias[col] (+ chained Cin)
        v8f acc[NT];
        #pragma unroll
        for (int nt = 0; nt < NT; ++nt) {
            int col  = nt * 16 + lr;
            float bb = bias ? bias[col] : 0.0f;
            #pragma unroll
            for (int r = 0; r < 8; ++r) {
                int rrow = mbase + r + halfsel * 8;
                int rr   = rrow < nrows ? rrow : nrows - 1;
                float c  = bb;
                if (Cin) c += Cin[(size_t)rr * NCOLS + col];
                acc[nt][r] = c;
            }
        }

        int arow = mbase + lr;
        if (arow >= nrows) arow = nrows - 1;
        const unsigned short* Arow = A + (size_t)arow * FDIM;

        // hoist ALL A fragments (4 K-steps x 2 x b128) into one load clause
        union frag_t { uint4 u[2]; v16bf v; };
        frag_t af[FDIM / 32];
        #pragma unroll
        for (int kt = 0; kt < FDIM / 32; ++kt) {
            const unsigned short* ap = Arow + kt * 32 + halfsel * 8;
            af[kt].u[0] = *(const uint4*)(ap);
            af[kt].u[1] = *(const uint4*)(ap + 16);
        }

        #pragma unroll
        for (int kt = 0; kt < FDIM / 32; ++kt) {
            // preload all B fragments for this K-step from LDS (lane = N col, elems = K in table order)
            frag_t bfr[NT];
            #pragma unroll
            for (int nt = 0; nt < NT; ++nt) {
                const unsigned short* bp = Bs + (size_t)(kt * 32 + halfsel * 8) * NCOLS + nt * 16 + lr;
                #pragma unroll
                for (int s = 0; s < 16; ++s)
                    ((unsigned short*)bfr[nt].u)[s] = bp[(size_t)((s & 7) + (s >> 3) * 16) * NCOLS];
            }
            #pragma unroll
            for (int nt = 0; nt < NT; ++nt) {
                acc[nt] = __builtin_amdgcn_wmma_f32_16x16x32_bf16(
                    false, af[kt].v, false, bfr[nt].v, (short)0, acc[nt], false, false);
            }
        }

        #pragma unroll
        for (int nt = 0; nt < NT; ++nt) {
            int col = nt * 16 + lr;
            #pragma unroll
            for (int r = 0; r < 8; ++r) {
                int rrow = mbase + r + halfsel * 8;
                if (rrow < nrows) {
                    float vv = acc[nt][r];
                    if (relu) vv = vv > 0.0f ? vv : 0.0f;
                    if (outF)  outF [(size_t)rrow * NCOLS + col] = vv;
                    if (outBf) outBf[(size_t)rrow * NCOLS + col] = f2bf(vv);
                }
            }
        }
    }
}

// ---------- edge pass 1: score + segment-max (flipped-uint atomicMax) ----------
__global__ void edge_score_max_kernel(const int* __restrict__ src, const int* __restrict__ dst,
                                      const float* __restrict__ s_src, const float* __restrict__ s_dst,
                                      float* __restrict__ e_edge, unsigned* __restrict__ emax, int E) {
    int e = blockIdx.x * blockDim.x + threadIdx.x;
    if (e >= E) return;
    float s  = s_src[src[e]] + s_dst[dst[e]];
    float es = s > 0.0f ? s : 0.2f * s;      // leaky_relu(0.2)
    e_edge[e] = es;
    atomicMax(emax + dst[e], fflip(es));
}

// ---------- edge pass 2: exp + segment-sum ----------
__global__ void edge_expsum_kernel(const int* __restrict__ dst, float* __restrict__ e_edge,
                                   const unsigned* __restrict__ emax, float* __restrict__ denom, int E) {
    int e = blockIdx.x * blockDim.x + threadIdx.x;
    if (e >= E) return;
    int d = dst[e];
    float ex = __expf(e_edge[e] - funflip(emax[d]));
    e_edge[e] = ex;                           // buffer now holds exp-score
    atomicAdd(denom + d, ex);
}

// ---------- edge pass 3: agg[dst] += m[src] * alpha  (wave per edge, float4/lane) ----------
__global__ void edge_scatter_kernel(const int* __restrict__ src, const int* __restrict__ dst,
                                    const float* __restrict__ e_edge, const float* __restrict__ denom,
                                    const float* __restrict__ m, float* __restrict__ agg, int E) {
    int edge = blockIdx.x * (blockDim.x >> 5) + (threadIdx.x >> 5);
    if (edge >= E) return;
    int lane = threadIdx.x & 31;
    int s = src[edge], d = dst[edge];
    float alpha = e_edge[edge] / (denom[d] + 1e-9f);
    const float4 mv = ((const float4*)(m + (size_t)s * FDIM))[lane];
    float* ap = agg + (size_t)d * FDIM + lane * 4;
    atomicAdd(ap + 0, mv.x * alpha);
    atomicAdd(ap + 1, mv.y * alpha);
    atomicAdd(ap + 2, mv.z * alpha);
    atomicAdd(ap + 3, mv.w * alpha);
}

// ---------- launch ----------
extern "C" void kernel_launch(void* const* d_in, const int* in_sizes, int n_in,
                              void* d_out, int out_size, void* d_ws, size_t ws_size,
                              hipStream_t stream) {
    const float* h    = (const float*)d_in[0];
    const int*   src  = (const int*)  d_in[1];
    const int*   dst  = (const int*)  d_in[2];
    const float* W1   = (const float*)d_in[3];
    const float* b1   = (const float*)d_in[4];
    const float* W2   = (const float*)d_in[5];
    const float* b2   = (const float*)d_in[6];
    const float* nk   = (const float*)d_in[7];
    const float* attn = (const float*)d_in[8];   // [2F] : a1 | a2
    const float* Wu   = (const float*)d_in[9];   // [2F,F] : top(agg) | bottom(h)
    const float* bu   = (const float*)d_in[10];
    const float* Wr1  = (const float*)d_in[11];  // [F,64]
    const float* br1  = (const float*)d_in[12];
    const float* Wr2  = (const float*)d_in[13];  // [64]
    const float* br2  = (const float*)d_in[14];
    float* out = (float*)d_out;

    const int N   = in_sizes[0] / FDIM;
    const int E   = in_sizes[1];
    const int HID = in_sizes[12];                // 64

    // carve workspace (all 256B aligned)
    char*  ws  = (char*)d_ws;
    size_t off = 0;
    auto carve = [&](size_t bytes) -> char* {
        char* p = ws + off;
        off += (bytes + 255) & ~(size_t)255;
        return p;
    };
    unsigned short* h_bf   = (unsigned short*)carve((size_t)N * FDIM * 2);
    unsigned short* t_bf   = (unsigned short*)carve((size_t)N * FDIM * 2);
    float*          m_f    = (float*)         carve((size_t)N * FDIM * 4);
    unsigned short* W1b    = (unsigned short*)carve((size_t)FDIM * FDIM * 2);
    unsigned short* W2b    = (unsigned short*)carve((size_t)FDIM * FDIM * 2);
    unsigned short* WuB    = (unsigned short*)carve((size_t)2 * FDIM * FDIM * 2);
    unsigned short* Wr1b   = (unsigned short*)carve((size_t)FDIM * HID * 2);
    float*          nka1   = (float*)         carve((size_t)FDIM * 4);
    float*          s_src  = (float*)         carve((size_t)N * 4);
    float*          s_dst  = (float*)         carve((size_t)N * 4);
    float*          e_edge = (float*)         carve((size_t)E * 4);
    unsigned*       emax   = (unsigned*)      carve((size_t)N * 4);
    float*          denom  = (float*)         carve((size_t)N * 4);
    float*          agg    = (float*)         carve((size_t)N * FDIM * 4);
    unsigned short* agg_bf = (unsigned short*)carve((size_t)N * FDIM * 2);
    float*          tmp    = (float*)         carve((size_t)N * FDIM * 4);
    unsigned short* hn_bf  = (unsigned short*)carve((size_t)N * FDIM * 2);
    float*          r1     = (float*)         carve((size_t)N * HID * 4);

    dim3 blk(256);
    // zero segment buffers
    init_kernel<<<(N * FDIM + 255) / 256, blk, 0, stream>>>(agg, denom, emax, N, N * FDIM);
    // f32 -> bf16 conversions
    cvt_kernel<<<(N * FDIM + 255) / 256, blk, 0, stream>>>(h, h_bf, N * FDIM);
    cvt_kernel<<<(FDIM * FDIM + 255) / 256, blk, 0, stream>>>(W1, W1b, FDIM * FDIM);
    cvt_kernel<<<(FDIM * FDIM + 255) / 256, blk, 0, stream>>>(W2, W2b, FDIM * FDIM);
    cvt_kernel<<<(2 * FDIM * FDIM + 255) / 256, blk, 0, stream>>>(Wu, WuB, 2 * FDIM * FDIM);
    cvt_kernel<<<(FDIM * HID + 255) / 256, blk, 0, stream>>>(Wr1, Wr1b, FDIM * HID);
    // fold node_kernel into attention: nka1 = node_kernel @ a1
    rowdot_kernel<<<(FDIM + 7) / 8, blk, 0, stream>>>(nk, attn, nullptr, nka1, FDIM, FDIM);

    // node-level WMMA GEMMs: persistent blocks, 4 waves/block, LDS-staged weights
    int tiles = (N + 15) / 16;
    dim3 gblk(128);
    int  ggrid = (tiles + 3) / 4;
    if (ggrid > 120) ggrid = 120;             // persistent: amortize LDS staging
    // t = relu(h @ W1 + b1)
    gemm_bf16_wmma_kernel<FDIM><<<ggrid, gblk, 0, stream>>>(h_bf, W1b, b1, nullptr, nullptr, t_bf, N, 1);
    // m = t @ W2 + b2   (per-node messages)
    gemm_bf16_wmma_kernel<FDIM><<<ggrid, gblk, 0, stream>>>(t_bf, W2b, b2, nullptr, m_f, nullptr, N, 0);
    // tmp = h @ Wu_bottom   (second half of the concat-update)
    gemm_bf16_wmma_kernel<FDIM><<<ggrid, gblk, 0, stream>>>(h_bf, WuB + FDIM * FDIM, nullptr, nullptr, tmp, nullptr, N, 0);

    // per-node attention scalars
    rowdot_kernel<<<(N + 7) / 8, blk, 0, stream>>>(m_f, nka1, nullptr, s_src, N, FDIM);
    rowdot_kernel<<<(N + 7) / 8, blk, 0, stream>>>(h, attn + FDIM, nullptr, s_dst, N, FDIM);

    // edge-level segment softmax + weighted scatter
    edge_score_max_kernel<<<(E + 255) / 256, blk, 0, stream>>>(src, dst, s_src, s_dst, e_edge, emax, E);
    edge_expsum_kernel<<<(E + 255) / 256, blk, 0, stream>>>(dst, e_edge, emax, denom, E);
    edge_scatter_kernel<<<(E + 7) / 8, blk, 0, stream>>>(src, dst, e_edge, denom, m_f, agg, E);

    // h_new = relu(agg @ Wu_top + tmp + bu)
    cvt_kernel<<<(N * FDIM + 255) / 256, blk, 0, stream>>>(agg, agg_bf, N * FDIM);
    gemm_bf16_wmma_kernel<FDIM><<<ggrid, gblk, 0, stream>>>(agg_bf, WuB, bu, tmp, nullptr, hn_bf, N, 1);
    // readout hidden: r1 = relu(h_new @ Wr1 + br1)   [N,64]
    gemm_bf16_wmma_kernel<64><<<ggrid, gblk, 0, stream>>>(hn_bf, Wr1b, br1, nullptr, r1, nullptr, N, 1);
    // out = r1 @ Wr2 + br2
    rowdot_kernel<<<(N + 7) / 8, blk, 0, stream>>>(r1, Wr2, br2, out, N, HID);

    (void)n_in; (void)out_size; (void)ws_size;
}
